// EnsemblePointNet_4380866642466
// MI455X (gfx1250) — matrix operand order repaired and checked by reading, not compiled, not run on hardware
//
#include <hip/hip_runtime.h>

typedef _Float16 v8h  __attribute__((ext_vector_type(8)));
typedef _Float16 v16h __attribute__((ext_vector_type(16)));
typedef float    v8f  __attribute__((ext_vector_type(8)));
typedef float    v4f  __attribute__((ext_vector_type(4)));

#define MMODELS 1296
#define BATCH   1024
#define DIN     8
#define HN      128
#define BT      128          // batch tile rows per iteration
#define NBT     (BATCH/BT)   // 8
#define WSTR    136          // f16 row stride (128 + 8 pad): 272B, 16B-aligned, bank-shifted
#define W0STR   40           // f16 row stride for W0^T (k 0..7 valid, 8..39 zero), 80B, 16B-aligned

// A-fragment for v_wmma_f32_16x16x32_f16 (row-major f16 source in LDS):
// halves j = A[row][kb + 16h + j] (16 contiguous halves starting at 16h)
__device__ __forceinline__ v16h load_afrag(const _Float16* rowp, int hh) {
  v8h lo = *(const v8h*)(rowp + 16*hh);
  v8h hi = *(const v8h*)(rowp + 16*hh + 8);
  v16h r;
  #pragma unroll
  for (int j = 0; j < 8; ++j) { r[j] = lo[j]; r[8+j] = hi[j]; }
  return r;
}

// B-fragment: halves j = B[kb + 16h + j][col]; with W stored transposed
// (colp = &Wt[col][kb]) this is 16 contiguous halves starting at 16h.
__device__ __forceinline__ v16h load_bfrag(const _Float16* colp, int hh) {
  v8h lo = *(const v8h*)(colp + 16*hh);
  v8h hi = *(const v8h*)(colp + 16*hh + 8);
  v16h r;
  #pragma unroll
  for (int j = 0; j < 8; ++j) { r[j] = lo[j]; r[8+j] = hi[j]; }
  return r;
}

// One 128x128 hidden layer for this wave's 16 rows: C = relu(A*W + b)
// Zero-init acc (first WMMA uses inline-0 C); bias+relu fused in epilogue.
__device__ __forceinline__ void layer128(const _Float16* bufIn,
                                         const _Float16* Wt,
                                         const float* bias,
                                         _Float16* bufOut,
                                         int rb, int l15, int hh)
{
  v16h af[4];
  #pragma unroll
  for (int kk = 0; kk < 4; ++kk)
    af[kk] = load_afrag(bufIn + (rb + l15)*WSTR + kk*32, hh);

  v8f acc[8] = {};
  #pragma unroll
  for (int kk = 0; kk < 4; ++kk) {
    v16h bf[8];
    #pragma unroll
    for (int tc = 0; tc < 8; ++tc)
      bf[tc] = load_bfrag(Wt + (tc*16 + l15)*WSTR + kk*32, hh);
    #pragma unroll
    for (int tc = 0; tc < 8; ++tc)
      acc[tc] = __builtin_amdgcn_wmma_f32_16x16x32_f16(
          false, af[kk], false, bf[tc], (short)0, acc[tc], false, false);
  }

  #pragma unroll
  for (int tc = 0; tc < 8; ++tc) {
    float bv = bias[tc*16 + l15];
    #pragma unroll
    for (int r = 0; r < 8; ++r) {
      float v = fmaxf(acc[tc][r] + bv, 0.f);
      bufOut[(rb + r + 8*hh)*WSTR + tc*16 + l15] = (_Float16)v;
    }
  }
}

__global__ __launch_bounds__(256, 2)
void ensemble_mlp_kernel(const float* __restrict__ xg,
                         const float* __restrict__ W0g, const float* __restrict__ b0g,
                         const float* __restrict__ W1g, const float* __restrict__ b1g,
                         const float* __restrict__ W2g, const float* __restrict__ b2g,
                         const float* __restrict__ W3g, const float* __restrict__ b3g,
                         float* __restrict__ outg)
{
  __shared__ __align__(16) _Float16 sW1t[HN * WSTR];   // W1^T f16   34816 B
  __shared__ __align__(16) _Float16 sW2t[HN * WSTR];   // W2^T f16   34816 B
  __shared__ __align__(16) _Float16 sW0t[HN * W0STR];  // W0^T, k>=8 zeroed, 10240 B
  __shared__ __align__(16) _Float16 sW3h[HN];          // W3 column f16
  __shared__ __align__(16) _Float16 bufA[BT * WSTR];   // activation ping 34816 B
  __shared__ __align__(16) _Float16 bufB[BT * WSTR];   // activation pong 34816 B
  __shared__ float sb0[HN], sb1[HN], sb2[HN];
  __shared__ float sb3;
  // total ~151.3 KB -> 2 blocks per WGP (302.6 KB of 320 KB), 4 waves/SIMD

  const int m    = blockIdx.x;
  const int t    = threadIdx.x;
  const int lane = t & 31;
  const int wv   = t >> 5;        // wave 0..7
  const int l15  = lane & 15;     // fragment column / row-in-tile
  const int hh   = lane >> 4;     // lane half
  const int rb   = wv * 16;       // this wave's row base in the batch tile

  // ---- stage weights to LDS (transposed, f32 -> f16), once per model ----
  const float* W1m = W1g + (size_t)m * HN * HN;
  const float* W2m = W2g + (size_t)m * HN * HN;
  for (int i = t; i < HN*HN; i += 256) {        // coalesced global reads
    int k = i >> 7, c = i & 127;
    sW1t[c*WSTR + k] = (_Float16)W1m[i];
    sW2t[c*WSTR + k] = (_Float16)W2m[i];
  }
  const float* W0m = W0g + (size_t)m * DIN * HN;
  for (int i = t; i < DIN*HN; i += 256) {
    int k = i >> 7, c = i & 127;                // k 0..7, c 0..127
    sW0t[c*W0STR + k] = (_Float16)W0m[i];
  }
  for (int i = t; i < HN*32; i += 256) {        // zero W0^T k = 8..39
    int c = i >> 5, k = 8 + (i & 31);
    sW0t[c*W0STR + k] = (_Float16)0.f;
  }
  if (t < HN) {
    sW3h[t] = (_Float16)W3g[(size_t)m*HN + t];
    sb0[t]  = b0g[(size_t)m*HN + t];
    sb1[t]  = b1g[(size_t)m*HN + t];
    sb2[t]  = b2g[(size_t)m*HN + t];
  }
  if (t == 0) sb3 = b3g[m];
  __syncthreads();  // only barrier: everything below is wave-local

  const _Float16 hz = (_Float16)0.f;

  // Layer-3 B fragments are pure weights -> hoist out of the batch loop.
  v16h w3f[4];
  #pragma unroll
  for (int kk = 0; kk < 4; ++kk)
    w3f[kk] = load_bfrag(sW3h + kk*32, hh);     // same values for every column

  for (int bt = 0; bt < NBT; ++bt) {
    // ---- layer 0: A built directly from global x (K padded 8->32 in regs).
    //      B fragments need no masking: A's k>=8 slots are zero and the
    //      W0^T LDS region for k=8..31 is zeroed, so 0*finite = 0. ----
    {
      const float* xp = xg + ((size_t)m*BATCH + bt*BT + rb + l15)*DIN;
      v4f x0 = *(const v4f*)xp;
      v4f x1 = *(const v4f*)(xp + 4);
      v16h a;
      a[0] = hh ? hz : (_Float16)x0.x;  a[1] = hh ? hz : (_Float16)x0.y;
      a[2] = hh ? hz : (_Float16)x0.z;  a[3] = hh ? hz : (_Float16)x0.w;
      a[4] = hh ? hz : (_Float16)x1.x;  a[5] = hh ? hz : (_Float16)x1.y;
      a[6] = hh ? hz : (_Float16)x1.z;  a[7] = hh ? hz : (_Float16)x1.w;
      #pragma unroll
      for (int j = 0; j < 8; ++j) a[8+j] = hz;

      v16h bf[8];
      #pragma unroll
      for (int tc = 0; tc < 8; ++tc)
        bf[tc] = load_bfrag(sW0t + (tc*16 + l15)*W0STR, hh);

      v8f acc[8] = {};
      #pragma unroll
      for (int tc = 0; tc < 8; ++tc)
        acc[tc] = __builtin_amdgcn_wmma_f32_16x16x32_f16(
            false, a, false, bf[tc], (short)0, acc[tc], false, false);

      #pragma unroll
      for (int tc = 0; tc < 8; ++tc) {
        float bv = sb0[tc*16 + l15];
        #pragma unroll
        for (int r = 0; r < 8; ++r) {
          float v = fmaxf(acc[tc][r] + bv, 0.f);
          bufA[(rb + r + 8*hh)*WSTR + tc*16 + l15] = (_Float16)v;
        }
      }
    }

    // ---- layers 1 & 2 ----
    layer128(bufA, sW1t, sb1, bufB, rb, l15, hh);
    layer128(bufB, sW2t, sb2, bufA, rb, l15, hh);

    // ---- layer 3: [16x128] x [128x1]; W3 broadcast to all 16 cols ----
    {
      v8f c = {};
      #pragma unroll
      for (int kk = 0; kk < 4; ++kk) {
        v16h a = load_afrag(bufA + (rb + l15)*WSTR + kk*32, hh);
        c = __builtin_amdgcn_wmma_f32_16x16x32_f16(
            false, a, false, w3f[kk], (short)0, c, false, false);
      }
      if (l15 == 0) {   // every D column equals the true output; take col 0
        #pragma unroll
        for (int r = 0; r < 8; ++r)
          outg[(size_t)m*BATCH + bt*BT + rb + r + 8*hh] = c[r] + sb3;
      }
    }
  }
}

extern "C" void kernel_launch(void* const* d_in, const int* in_sizes, int n_in,
                              void* d_out, int out_size, void* d_ws, size_t ws_size,
                              hipStream_t stream) {
  const float* x  = (const float*)d_in[0];
  const float* W0 = (const float*)d_in[1];
  const float* b0 = (const float*)d_in[2];
  const float* W1 = (const float*)d_in[3];
  const float* b1 = (const float*)d_in[4];
  const float* W2 = (const float*)d_in[5];
  const float* b2 = (const float*)d_in[6];
  const float* W3 = (const float*)d_in[7];
  const float* b3 = (const float*)d_in[8];
  float* out = (float*)d_out;
  dim3 grid(MMODELS), block(256);
  hipLaunchKernelGGL(ensemble_mlp_kernel, grid, block, 0, stream,
                     x, W0, b0, W1, b1, W2, b2, W3, b3, out);
}
